// MultiHeadAttentionQuantum_65481071402934
// MI455X (gfx1250) — matrix operand order repaired and checked by reading, not compiled, not run on hardware
//
#include <hip/hip_runtime.h>
#include <hip/hip_bf16.h>
#include <math.h>

typedef __attribute__((ext_vector_type(16))) _Float16 v16h;
typedef __attribute__((ext_vector_type(8)))  _Float16 v8h;
typedef __attribute__((ext_vector_type(8)))  float    v8f;

#define EMBED   768
#define NQ      96
#define HEADS   12
#define DK      8
#define BATCH   8
#define SEQ     1024
#define GHEADS  (BATCH*HEADS)     // 96 (b,h) attention problems

// Fold both 1/sqrt(d_k) and log2(e) into Q and K:
//   scores_for_exp2 = (q·q)/sqrt(8) * log2(e)  ==>  scale q by 8^(-1/4)*sqrt(log2 e)
// |q_scaled| <= 0.7142 -> |score| <= 4.08 -> exp2 in [0.059, 17]: always inside
// f16/f32 range, so NO running max / online rescale is needed (softmax is
// shift-invariant; with bounded scores the shift can be zero).
#define QSCALE  0.7141916f        // 8^(-1/4) * sqrt(log2(e))

// ---------------------------------------------------------------------------
// Kernel 1: q[b,s,n] = cos(x[b,s,n])*cos(theta[n]), n < 96.
//   qrow[g][s][8]  (scaled; Q and K fragments, one b128 per row)
//   qT  [g][8][s]  (unscaled; V in B-matrix layout, contiguous keys)
// ---------------------------------------------------------------------------
__global__ __launch_bounds__(256)
void qgen_kernel(const float* __restrict__ x, const float* __restrict__ theta,
                 _Float16* __restrict__ qrow, _Float16* __restrict__ qT) {
  int idx = blockIdx.x * blockDim.x + threadIdx.x;
  if (idx >= BATCH * SEQ * NQ) return;
  int n   = idx % NQ;
  int row = idx / NQ;                 // b*SEQ + s
  int b = row / SEQ, s = row % SEQ;
  int h = n / DK,  d = n % DK;
  int g = b * HEADS + h;
  float v = __cosf(x[row * EMBED + n]) * __cosf(theta[n]);
  qrow[(g * SEQ + s) * DK + d] = (_Float16)(v * QSCALE);
  qT[(g * DK + d) * SEQ + s]   = (_Float16)v;
}

// ---------------------------------------------------------------------------
// Kernel 2: W (768x96 f32, row-major) -> f16
// ---------------------------------------------------------------------------
__global__ __launch_bounds__(256)
void wcvt_kernel(const float* __restrict__ W, _Float16* __restrict__ W16) {
  int i = blockIdx.x * blockDim.x + threadIdx.x;
  if (i < EMBED * NQ) W16[i] = (_Float16)W[i];
}

// ---------------------------------------------------------------------------
// Kernel 3: attention per head. One workgroup = one head x 256 queries
// (8 waves, TWO 16-row query tiles per wave so each K/V fragment load feeds
// 4 score WMMAs + 2 PV WMMAs). Full head K/V staged in LDS with appended
// ZERO rows so padding lanes do unconditional loads (no EXEC divergence).
// Fixed-shift softmax: P = exp2(S), per-lane partial row sums, one cross-lane
// reduction at the end.
// ---------------------------------------------------------------------------
__global__ __launch_bounds__(256)
void flash_kernel(const _Float16* __restrict__ qrow, const _Float16* __restrict__ qT,
                  _Float16* __restrict__ attn16) {
  __shared__ __align__(16) _Float16 sQ[(SEQ + 16) * DK]; // 16.25 KB; rows SEQ.. zero
  __shared__ __align__(16) _Float16 sKT[9 * SEQ];        // 18 KB; row 8 = zeros
  __shared__ __align__(16) _Float16 sP[8][1024];         // 16 KB; per-wave P bounce

  const int g    = blockIdx.x >> 2;      // head index (0..95)
  const int qb   = blockIdx.x & 3;       // 256-query block
  const int tid  = threadIdx.x;
  const int lane = tid & 31;
  const int wave = tid >> 5;
  const int half = lane >> 4;            // 0: lanes 0-15, 1: lanes 16-31
  const int l16  = lane & 15;

  // Cooperative stage of the head's K/V into LDS + zero pad rows.
  {
    const v8h* gq = (const v8h*)(qrow + (size_t)g * SEQ * DK);
    const v8h* gk = (const v8h*)(qT   + (size_t)g * DK * SEQ);
    v8h* lq = (v8h*)sQ;
    v8h* lk = (v8h*)sKT;
    for (int i = tid; i < SEQ * DK / 8; i += 256) { lq[i] = gq[i]; lk[i] = gk[i]; }
    v8h z = {};
    if (tid < 16) lq[SEQ * DK / 8 + tid] = z;            // 16 zero q-rows
    for (int i = tid; i < SEQ / 8; i += 256) lk[DK * SEQ / 8 + i] = z; // zero V row
  }
  __syncthreads();

  const int q0 = qb * 256 + wave * 32;   // first query row of this wave

  // A-matrix fragments for two Q tiles (16x32, K = d padded 8->32 with zeros).
  // Lanes 16-31 (K=8..15) read the zero rows; elements 8-15 stay zero.
  const int zrow  = SEQ + l16;
  const int rowqA = half ? zrow : (q0 + l16);
  const int rowqB = half ? zrow : (q0 + 16 + l16);
  v16h aqA = {}, aqB = {};
  {
    v8h qa = *(const v8h*)(sQ + rowqA * DK);
    v8h qb8 = *(const v8h*)(sQ + rowqB * DK);
    #pragma unroll
    for (int i = 0; i < 8; ++i) { aqA[i] = qa[i]; aqB[i] = qb8[i]; }
  }
  const int rowv = (l16 < DK) ? l16 : DK;   // d>=8 -> zero row of sKT

  v8f oA = {}, oB = {};        // unnormalized PV accumulators
  v8f lsumA = {}, lsumB = {};  // per-lane partial row sums of P

  _Float16* pbuf = &sP[wave][0];

  for (int kb = 0; kb < SEQ; kb += 32) {
    // B-matrix fragments for two 16-key tiles (32x16, K = d padded).
    const int rowk0 = half ? zrow : (kb + l16);
    const int rowk1 = half ? zrow : (kb + 16 + l16);
    v16h bk0 = {}, bk1 = {};
    {
      v8h k0 = *(const v8h*)(sQ + rowk0 * DK);
      v8h k1 = *(const v8h*)(sQ + rowk1 * DK);
      #pragma unroll
      for (int i = 0; i < 8; ++i) { bk0[i] = k0[i]; bk1[i] = k1[i]; }
    }
    v8f z8 = {};
    v8f c0a = __builtin_amdgcn_wmma_f32_16x16x32_f16(false, aqA, false, bk0, (short)0, z8, false, false);
    v8f c1a = __builtin_amdgcn_wmma_f32_16x16x32_f16(false, aqA, false, bk1, (short)0, z8, false, false);
    v8f c0b = __builtin_amdgcn_wmma_f32_16x16x32_f16(false, aqB, false, bk0, (short)0, z8, false, false);
    v8f c1b = __builtin_amdgcn_wmma_f32_16x16x32_f16(false, aqB, false, bk1, (short)0, z8, false, false);

    // P = exp2(S); accumulate per-lane row sums; bounce P (D-layout: lane =
    // key col) to LDS row-major [query][key-in-block] for both tiles.
    #pragma unroll
    for (int r = 0; r < 8; ++r) {
      float p0a = __builtin_amdgcn_exp2f(c0a[r]);
      float p1a = __builtin_amdgcn_exp2f(c1a[r]);
      float p0b = __builtin_amdgcn_exp2f(c0b[r]);
      float p1b = __builtin_amdgcn_exp2f(c1b[r]);
      lsumA[r] += p0a + p1a;
      lsumB[r] += p0b + p1b;
      int pr = (r + 8 * half) * 32 + l16;
      pbuf[pr]            = (_Float16)p0a;
      pbuf[pr + 16]       = (_Float16)p1a;
      pbuf[512 + pr]      = (_Float16)p0b;
      pbuf[512 + pr + 16] = (_Float16)p1b;
    }

    // Reload P as A-matrix fragments (16x32 f16): lane m = query, K = key.
    v16h apA, apB;
    {
      const int koff = l16 * 32 + (half ? 8 : 0);
      v8h alo = *(const v8h*)(pbuf + koff);
      v8h ahi = *(const v8h*)(pbuf + koff + 16);
      v8h blo = *(const v8h*)(pbuf + 512 + koff);
      v8h bhi = *(const v8h*)(pbuf + 512 + koff + 16);
      #pragma unroll
      for (int i = 0; i < 8; ++i) {
        apA[i] = alo[i]; apA[8 + i] = ahi[i];
        apB[i] = blo[i]; apB[8 + i] = bhi[i];
      }
    }

    // B-matrix fragment for V (32x16: K = 32 keys, N = d padded 8->16 via
    // the zero row of sKT).
    v16h bv;
    {
      const v8h* vp = (const v8h*)(sKT + rowv * SEQ + kb + 8 * half);
      v8h v0 = vp[0], v1 = vp[2];
      #pragma unroll
      for (int i = 0; i < 8; ++i) { bv[i] = v0[i]; bv[8 + i] = v1[i]; }
    }
    oA = __builtin_amdgcn_wmma_f32_16x16x32_f16(false, apA, false, bv, (short)0, oA, false, false);
    oB = __builtin_amdgcn_wmma_f32_16x16x32_f16(false, apB, false, bv, (short)0, oB, false, false);
  }

  // Cross-lane row-sum reductions (16-lane halves), normalize, write attn
  // (f16) as [b*S+s][96], col = h*8+d.
  const int brow0 = (g / HEADS) * SEQ;
  const int h = g % HEADS;
  #pragma unroll
  for (int r = 0; r < 8; ++r) {
    float lsA = lsumA[r];
    lsA += __shfl_xor(lsA, 1);
    lsA += __shfl_xor(lsA, 2);
    lsA += __shfl_xor(lsA, 4);
    lsA += __shfl_xor(lsA, 8);
    float lsB = lsumB[r];
    lsB += __shfl_xor(lsB, 1);
    lsB += __shfl_xor(lsB, 2);
    lsB += __shfl_xor(lsB, 4);
    lsB += __shfl_xor(lsB, 8);
    if (l16 < DK) {
      int srow = q0 + r + 8 * half;
      attn16[(size_t)(brow0 + srow) * NQ + h * DK + l16]      = (_Float16)(oA[r] / lsA);
      attn16[(size_t)(brow0 + srow + 16) * NQ + h * DK + l16] = (_Float16)(oB[r] / lsB);
    }
  }
}

// ---------------------------------------------------------------------------
// Kernel 4: out(8192x768) = attn16(8192x96) x W16^T(96x768), K = 96 = 3x32.
// One wave per 16x16 output tile, 8 waves per block.
// ---------------------------------------------------------------------------
__global__ __launch_bounds__(256)
void proj_kernel(const _Float16* __restrict__ attn16, const _Float16* __restrict__ W16,
                 float* __restrict__ out) {
  const int lane = threadIdx.x & 31;
  const int wave = threadIdx.x >> 5;
  const int half = lane >> 4, l16 = lane & 15;
  const int COLT = EMBED / 16;                    // 48
  const int tile = blockIdx.x * 8 + wave;         // 512*48 = 24576 tiles exactly
  const int rt = tile / COLT, ct = tile % COLT;

  const int arow = rt * 16 + l16;                 // attn row for A fragment
  const int ecol = ct * 16 + l16;                 // W row (= output column)
  v8f acc = {};
  #pragma unroll
  for (int kc = 0; kc < 3; ++kc) {
    const int k0 = kc * 32;
    v16h a, b;
    {
      const _Float16* ap = attn16 + (size_t)arow * NQ + k0 + (half ? 8 : 0);
      v8h lo = *(const v8h*)ap;
      v8h hi = *(const v8h*)(ap + 16);
      #pragma unroll
      for (int i = 0; i < 8; ++i) { a[i] = lo[i]; a[8 + i] = hi[i]; }
    }
    {
      const _Float16* bp = W16 + (size_t)ecol * NQ + k0 + 8 * half;
      v8h lo = *(const v8h*)bp;
      v8h hi = *(const v8h*)(bp + 16);
      #pragma unroll
      for (int i = 0; i < 8; ++i) { b[i] = lo[i]; b[8 + i] = hi[i]; }
    }
    acc = __builtin_amdgcn_wmma_f32_16x16x32_f16(false, a, false, b, (short)0, acc, false, false);
  }
  #pragma unroll
  for (int r = 0; r < 8; ++r)
    out[(size_t)(rt * 16 + r + 8 * half) * EMBED + ecol] = acc[r];
}

// ---------------------------------------------------------------------------
extern "C" void kernel_launch(void* const* d_in, const int* in_sizes, int n_in,
                              void* d_out, int out_size, void* d_ws, size_t ws_size,
                              hipStream_t stream) {
  const float* x     = (const float*)d_in[0];   // (8,1024,768)
  const float* theta = (const float*)d_in[1];   // (96,)
  const float* W     = (const float*)d_in[2];   // (768,96)
  float* out = (float*)d_out;                   // (8,1024,768)

  char* ws = (char*)d_ws;
  const size_t SZQ = (size_t)GHEADS * SEQ * DK * sizeof(_Float16);   // 1.5 MB
  _Float16* qrow   = (_Float16*)(ws);
  _Float16* qT     = (_Float16*)(ws + SZQ);
  _Float16* attn16 = (_Float16*)(ws + 2 * SZQ);
  _Float16* W16    = (_Float16*)(ws + 3 * SZQ);

  qgen_kernel<<<(BATCH * SEQ * NQ + 255) / 256, 256, 0, stream>>>(x, theta, qrow, qT);
  wcvt_kernel<<<(EMBED * NQ + 255) / 256, 256, 0, stream>>>(W, W16);
  flash_kernel<<<GHEADS * (SEQ / 256), 256, 0, stream>>>(qrow, qT, attn16);
  proj_kernel<<<(BATCH * SEQ / 16) * (EMBED / 16) / 8, 256, 0, stream>>>(attn16, W16, out);
}
